// gconv_73684458930377
// MI455X (gfx1250) — matrix operand order repaired and checked by reading, not compiled, not run on hardware
//
#include <hip/hip_runtime.h>

// Problem constants (match reference)
#define NN     10000      // num nodes
#define BB     8          // batch
#define INF    64         // input features per node
#define UNITS  64         // output units
#define KTOT   192        // IN * (K+1) Chebyshev-expanded features
#define KROW   1536       // KTOT * BB : floats per node in staged XS
#define MROWS  80000      // B*N rows of the dense GEMM
#define NKT    6          // K chunks of 32
#define WSWZ_HALFS (KTOT * UNITS)   // 12288 f16 = 24 KB

typedef __attribute__((ext_vector_type(16))) _Float16     v16h;
typedef __attribute__((ext_vector_type(8)))  float        v8f;
typedef __attribute__((ext_vector_type(4)))  unsigned int ui32x4;
typedef __attribute__((ext_vector_type(8)))  int          i32x8;
typedef __attribute__((ext_vector_type(4)))  int          i32x4;

// x0[node, i*8+b] == inputs[b, node*64 + i]  (index permutation of the input)
__device__ __forceinline__ float x0_gather(const float* __restrict__ inputs,
                                           int node, int i, int b) {
  return inputs[(size_t)b * (NN * INF) + (size_t)node * INF + i];
}

// Staged layout, linear in GEMM k (k = i*3 + mc):
//   XS[node*1536 + i*24 + mc*8 + b] == xs[mc][node, i*8 + b]
__global__ void init_xs_kernel(const float* __restrict__ inputs, float* __restrict__ XS) {
  int idx = blockIdx.x * blockDim.x + threadIdx.x;      // over N*IN*B = 5.12M
  if (idx >= NN * INF * BB) return;
  int node = idx >> 9;
  int i    = (idx >> 3) & 63;
  int b    = idx & 7;
  float v  = x0_gather(inputs, node, i, b);
  float* p = XS + (size_t)node * KROW + i * 24 + b;
  p[0]  = v;       // x0 slot
  p[8]  = 0.0f;    // x1 accumulator
  p[16] = -v;      // x2 = -x0 (+ 2*L*x1 added later)
}

// Pre-swizzle weights (once) into f16 WMMA B-register image:
//   WSWZ[((kt*4 + nt)*32 + lane)*16 + e] = f16(W[(kt*32 + (lane>>4)*8 + e + (e>=8 ? 8 : 0))*64 + nt*16 + (lane&15)])
__global__ void wswz_kernel(const float* __restrict__ W, unsigned short* __restrict__ WSWZ) {
  int idx = blockIdx.x * blockDim.x + threadIdx.x;      // 12288
  if (idx >= WSWZ_HALFS) return;
  int e    = idx & 15;
  int lane = (idx >> 4) & 31;
  int nt   = (idx >> 9) & 3;
  int kt   = idx >> 11;
  int k    = kt * 32 + (lane >> 4) * 8 + e + (e >= 8 ? 8 : 0);
  int u    = nt * 16 + (lane & 15);
  _Float16 h = (_Float16)W[k * UNITS + u];
  WSWZ[idx] = *(const unsigned short*)&h;
}

// x1 += vals[e] * x0[cols[e], :]   (one wave per nonzero, 512-wide row)
__global__ void spmm_x1_kernel(float* XS,
                               const int* __restrict__ rows, const int* __restrict__ cols,
                               const float* __restrict__ vals, int nnz) {
  int e = blockIdx.x * 8 + (threadIdx.x >> 5);
  if (e >= nnz) return;
  int lane = threadIdx.x & 31;
  int r = rows[e], c = cols[e];
  float v = vals[e];
  const float* src = XS + (size_t)c * KROW;       // x0 slots: +i*24 + b
  float*       dst = XS + (size_t)r * KROW + 8;   // x1 slots
#pragma unroll
  for (int t = 0; t < 16; ++t) {
    int j   = t * 32 + lane;
    int off = (j >> 3) * 24 + (j & 7);
    atomicAdd(dst + off, v * src[off]);           // resolves in 192MB L2
  }
}

// x2 += 2*vals[e] * x1[cols[e], :]
__global__ void spmm_x2_kernel(float* XS,
                               const int* __restrict__ rows, const int* __restrict__ cols,
                               const float* __restrict__ vals, int nnz) {
  int e = blockIdx.x * 8 + (threadIdx.x >> 5);
  if (e >= nnz) return;
  int lane = threadIdx.x & 31;
  int r = rows[e], c = cols[e];
  float v2 = 2.0f * vals[e];
  const float* src = XS + (size_t)c * KROW + 8;   // x1 slots
  float*       dst = XS + (size_t)r * KROW + 16;  // x2 slots
#pragma unroll
  for (int t = 0; t < 16; ++t) {
    int j   = t * 32 + lane;
    int off = (j >> 3) * 24 + (j & 7);
    atomicAdd(dst + off, v2 * src[off]);
  }
}

// Dense GEMM: out[m,u] = sum_k XS[node*1536 + k*8 + b] * W[k,u] + bias[u]
// Block = 128 threads = 4 waves; each wave owns one 16-row M-tile and the
// full 64-col output (4 accumulators). B comes from LDS (TDM-staged, f16,
// pre-swizzled to register layout) -> two ds_load_b128 per tile.
// NN = 625*16, so every 16-row tile has a uniform batch index b.
__global__ void __launch_bounds__(128)
gemm_wmma_kernel(const float* __restrict__ XS, const unsigned short* __restrict__ WSWZ,
                 const float* __restrict__ bias, float* __restrict__ out) {
  __shared__ unsigned short lds_w[WSWZ_HALFS];    // 24 KB

  // ---- stage swizzled W into LDS (one wave issues the TDM op) ----
#if __has_builtin(__builtin_amdgcn_tensor_load_to_lds) && __has_builtin(__builtin_amdgcn_s_wait_tensorcnt)
  if (threadIdx.x < 32) {
    unsigned long long ga = (unsigned long long)(uintptr_t)WSWZ;
    unsigned lds_off = (unsigned)(uintptr_t)&lds_w[0];          // low 32 bits = LDS byte offset
    ui32x4 g0 = { 1u,                                           // count=1 (valid user D#)
                  lds_off,                                      // lds_addr
                  (unsigned)ga,                                 // global_addr[31:0]
                  ((unsigned)(ga >> 32) & 0x01FFFFFFu) | 0x80000000u }; // addr[56:32] | type=2
    i32x8 g1 = { 0x00020000,   // data_size=2 (4-byte elems), no multicast/pad
                 0x18000000,   // tensor_dim0[15:0]=6144 in [31:16]
                 0x00010000,   // tensor_dim0[31:16]=0, tensor_dim1=1
                 0x18000000,   // tile_dim0=6144 in [31:16]
                 0x00000001,   // tile_dim1=1, tile_dim2=0
                 0x00001800,   // tensor_dim0_stride=6144
                 0, 0 };
    i32x4 z4 = { 0, 0, 0, 0 };
    i32x8 z8 = { 0, 0, 0, 0, 0, 0, 0, 0 };
    __builtin_amdgcn_tensor_load_to_lds(g0, g1, z4, z4, z8, 0);
    __builtin_amdgcn_s_wait_tensorcnt(0);
  }
#else
  {
    const uint4* src = (const uint4*)WSWZ;
    uint4*       dst = (uint4*)lds_w;
    for (int t = threadIdx.x; t < (int)(WSWZ_HALFS * 2 / 16); t += 128) dst[t] = src[t];
  }
#endif
  __syncthreads();

  const int lane = threadIdx.x & 31;
  const int wv   = threadIdx.x >> 5;
  const int mt   = blockIdx.x * 4 + wv;          // 0..4999
  const int l15  = lane & 15;
  const int hi   = lane >> 4;

  const int row0 = mt * 16;
  const int b    = row0 / NN;                    // uniform across the tile
  const int node = row0 - b * NN + l15;          // consecutive across lanes
  const float* abase = XS + (size_t)node * KROW + b + (size_t)hi * 64;
  const v16h*  bl    = (const v16h*)lds_w;       // [ (kt*4+nt)*32 + lane ]

  v8f acc0 = {}, acc1 = {}, acc2 = {}, acc3 = {};
#pragma unroll
  for (int kt = 0; kt < NKT; ++kt) {
    const int kbase = kt * 32;
    v16h a;
#pragma unroll
    for (int e = 0; e < 16; ++e) {
      int koff = kbase + e + (e >= 8 ? 8 : 0);   // compile-time; hi folded into abase
      a[e] = (_Float16)abase[(size_t)koff * 8];
    }
    v16h b0 = bl[(kt * 4 + 0) * 32 + lane];
    v16h b1 = bl[(kt * 4 + 1) * 32 + lane];
    v16h b2 = bl[(kt * 4 + 2) * 32 + lane];
    v16h b3 = bl[(kt * 4 + 3) * 32 + lane];
    acc0 = __builtin_amdgcn_wmma_f32_16x16x32_f16(false, a, false, b0, (short)0, acc0, false, false);
    acc1 = __builtin_amdgcn_wmma_f32_16x16x32_f16(false, a, false, b1, (short)0, acc1, false, false);
    acc2 = __builtin_amdgcn_wmma_f32_16x16x32_f16(false, a, false, b2, (short)0, acc2, false, false);
    acc3 = __builtin_amdgcn_wmma_f32_16x16x32_f16(false, a, false, b3, (short)0, acc3, false, false);
  }

#pragma unroll
  for (int j = 0; j < 8; ++j) {                  // C/D: VGPR j -> M = 8*hi + j, N = l15
    const size_t row = (size_t)(row0 + hi * 8 + j);
    out[row * UNITS +  0 + l15] = acc0[j] + bias[ 0 + l15];
    out[row * UNITS + 16 + l15] = acc1[j] + bias[16 + l15];
    out[row * UNITS + 32 + l15] = acc2[j] + bias[32 + l15];
    out[row * UNITS + 48 + l15] = acc3[j] + bias[48 + l15];
  }
}

extern "C" void kernel_launch(void* const* d_in, const int* in_sizes, int n_in,
                              void* d_out, int out_size, void* d_ws, size_t ws_size,
                              hipStream_t stream) {
  (void)n_in; (void)out_size; (void)ws_size;
  const float* inputs = (const float*)d_in[0];
  const int*   rows   = (const int*)d_in[1];
  const int*   cols   = (const int*)d_in[2];
  const float* vals   = (const float*)d_in[3];
  const float* W      = (const float*)d_in[4];
  const float* bias   = (const float*)d_in[5];
  float*       out    = (float*)d_out;
  const int    nnz    = in_sizes[1];             // 330000

  float*          XS   = (float*)d_ws;                           // 61.44 MB staged features
  unsigned short* WSWZ = (unsigned short*)((char*)d_ws + (size_t)NN * KROW * 4); // 24 KB

  const int total = NN * INF * BB;               // 5,120,000
  init_xs_kernel<<<(total + 255) / 256, 256, 0, stream>>>(inputs, XS);
  wswz_kernel<<<(WSWZ_HALFS + 255) / 256, 256, 0, stream>>>(W, WSWZ);
  spmm_x1_kernel<<<(nnz + 7) / 8, 256, 0, stream>>>(XS, rows, cols, vals, nnz);
  spmm_x2_kernel<<<(nnz + 7) / 8, 256, 0, stream>>>(XS, rows, cols, vals, nnz);
  gemm_wmma_kernel<<<MROWS / 16 / 4, 128, 0, stream>>>(XS, WSWZ, bias, out);
}
